// GCNHierarchicalClassifier_88072599372564
// MI455X (gfx1250) — compile-verified, matrix operand order checked by reading, not run on hardware
//
#include <hip/hip_runtime.h>

// Problem constants (match reference)
#define N0c 30000     // multiple of 16
#define F0c 128
#define Hc  256
#define Cc  16
#define N1c 3000
#define N1p 3008      // N1 padded to multiple of 16 for guard-free WMMA
#define Bc  16
#define E0c 960000
#define E1c 48000

typedef __attribute__((ext_vector_type(16))) _Float16 v16h;
typedef __attribute__((ext_vector_type(8)))  float    v8f;

// ---------------------------------------------------------------------------
// degree -> dinv  (gcn_norm with add_self_loops=True: deg = indeg + 1 > 0)
// ---------------------------------------------------------------------------
__global__ void deg_count(const int* __restrict__ dst, float* __restrict__ deg, int E) {
  int e = blockIdx.x * blockDim.x + threadIdx.x;
  if (e < E) atomicAdd(&deg[dst[e]], 1.0f);
}

__global__ void deg_to_dinv(float* __restrict__ deg, int n) {
  int i = blockIdx.x * blockDim.x + threadIdx.x;
  if (i < n) deg[i] = rsqrtf(deg[i] + 1.0f);  // +1 self loop, always > 0
}

// ---------------------------------------------------------------------------
// Pre-swizzle W[K,N] (f32, row-major) into the WMMA B-operand lane layout as
// f16.  Element order == exactly what each GEMM lane loads contiguously:
//   out[((tn*ksteps + ks)*32 + lane)*16 + j]
//     = W[ks*32 + kb(lane) + j + (j>=8 ? 8 : 0)][tn*16 + (lane&15)]
// Rows kk >= Kreal are zero (K padding, e.g. 17 -> 32).
// ---------------------------------------------------------------------------
__global__ void swizzle_w(const float* __restrict__ W, _Float16* __restrict__ out,
                          int Kp, int Kreal, int N) {
  int tid = blockIdx.x * blockDim.x + threadIdx.x;
  int ksteps = Kp >> 5;
  int total = (N >> 4) * ksteps * 512;
  if (tid >= total) return;
  int j    = tid & 15;
  int lane = (tid >> 4) & 31;
  int rest = tid >> 9;
  int ks   = rest % ksteps;
  int tn   = rest / ksteps;
  int kb   = (lane >> 4) << 3;
  int kk   = (ks << 5) + kb + j + ((j >= 8) ? 8 : 0);
  int col  = (tn << 4) + (lane & 15);
  float val = (kk < Kreal) ? W[(size_t)kk * N + col] : 0.0f;
  out[tid] = (_Float16)val;
}

// ---------------------------------------------------------------------------
// C[M,N] = A[M,Kp] @ B[Kp,N] via v_wmma_f32_16x16x32_f16, f32 accumulate.
// Guard-free, software-pipelined (depth 2): loads for step ks+1 are issued
// into fresh registers before the 4 WMMAs of step ks execute, so the
// s_wait_loadcnt for each fragment lands a full iteration of cvt+WMMA work
// after its loads, and WMMA->load WAR hazard NOPs disappear.
// One wave -> 16x64 output strip (4 tiles / 4 accumulators).
// ---------------------------------------------------------------------------
__global__ __launch_bounds__(256) void gemm_f16_wmma(
    const float* __restrict__ A, const _Float16* __restrict__ Bswz,
    float* __restrict__ C, int M, int Kp, int N) {
  const int lane    = threadIdx.x & 31;
  const int wid     = (blockIdx.x * blockDim.x + threadIdx.x) >> 5;
  const int nGroups = N >> 6;                 // groups of 4 N-tiles
  const int tm      = wid / nGroups;
  if (tm * 16 >= M) return;                   // wave-uniform -> EXEC stays full
  const int tg   = wid - tm * nGroups;
  const int half = lane >> 4;                 // 0: lanes 0-15, 1: lanes 16-31
  const int l15  = lane & 15;
  const int row  = tm * 16 + l15;
  const int kb   = half << 3;
  const int ksteps = Kp >> 5;

  const float*    Arow  = A + (size_t)row * Kp + kb;
  const _Float16* Bbase = Bswz + ((size_t)(tg * 4) * ksteps * 32 + (size_t)lane) * 16;
  const size_t    Btile = (size_t)ksteps * 512;   // f16 elements per N-tile

  v8f acc0 = {}, acc1 = {}, acc2 = {}, acc3 = {};

  // ---- prologue: load k-step 0 ----
  const float4* ap = (const float4*)(Arow);
  float4 A0 = ap[0], A1 = ap[1], A2 = ap[4], A3 = ap[5];
  v16h B0 = *(const v16h*)(Bbase);
  v16h B1 = *(const v16h*)(Bbase + Btile);
  v16h B2 = *(const v16h*)(Bbase + 2 * Btile);
  v16h B3 = *(const v16h*)(Bbase + 3 * Btile);

  for (int ks = 0; ks + 1 < ksteps; ++ks) {
    // current fragments (SSA copies; no real moves)
    float4 c0 = A0, c1 = A1, c2 = A2, c3 = A3;
    v16h d0 = B0, d1 = B1, d2 = B2, d3 = B3;

    // issue next iteration's loads into fresh registers
    const float4* an = (const float4*)(Arow + ((ks + 1) << 5));
    A0 = an[0]; A1 = an[1]; A2 = an[4]; A3 = an[5];
    const _Float16* bn = Bbase + (size_t)(ks + 1) * 512;
    B0 = *(const v16h*)(bn);
    B1 = *(const v16h*)(bn + Btile);
    B2 = *(const v16h*)(bn + 2 * Btile);
    B3 = *(const v16h*)(bn + 3 * Btile);
    __builtin_prefetch(Arow + ((ks + 2) << 5), 0, 1);

    // compute with current fragments
    v16h a;
    a[0]  = (_Float16)c0.x; a[1]  = (_Float16)c0.y;
    a[2]  = (_Float16)c0.z; a[3]  = (_Float16)c0.w;
    a[4]  = (_Float16)c1.x; a[5]  = (_Float16)c1.y;
    a[6]  = (_Float16)c1.z; a[7]  = (_Float16)c1.w;
    a[8]  = (_Float16)c2.x; a[9]  = (_Float16)c2.y;
    a[10] = (_Float16)c2.z; a[11] = (_Float16)c2.w;
    a[12] = (_Float16)c3.x; a[13] = (_Float16)c3.y;
    a[14] = (_Float16)c3.z; a[15] = (_Float16)c3.w;

    acc0 = __builtin_amdgcn_wmma_f32_16x16x32_f16(false, a, false, d0, (short)0, acc0, false, false);
    acc1 = __builtin_amdgcn_wmma_f32_16x16x32_f16(false, a, false, d1, (short)0, acc1, false, false);
    acc2 = __builtin_amdgcn_wmma_f32_16x16x32_f16(false, a, false, d2, (short)0, acc2, false, false);
    acc3 = __builtin_amdgcn_wmma_f32_16x16x32_f16(false, a, false, d3, (short)0, acc3, false, false);
  }

  // ---- epilogue: last k-step ----
  {
    v16h a;
    a[0]  = (_Float16)A0.x; a[1]  = (_Float16)A0.y;
    a[2]  = (_Float16)A0.z; a[3]  = (_Float16)A0.w;
    a[4]  = (_Float16)A1.x; a[5]  = (_Float16)A1.y;
    a[6]  = (_Float16)A1.z; a[7]  = (_Float16)A1.w;
    a[8]  = (_Float16)A2.x; a[9]  = (_Float16)A2.y;
    a[10] = (_Float16)A2.z; a[11] = (_Float16)A2.w;
    a[12] = (_Float16)A3.x; a[13] = (_Float16)A3.y;
    a[14] = (_Float16)A3.z; a[15] = (_Float16)A3.w;
    acc0 = __builtin_amdgcn_wmma_f32_16x16x32_f16(false, a, false, B0, (short)0, acc0, false, false);
    acc1 = __builtin_amdgcn_wmma_f32_16x16x32_f16(false, a, false, B1, (short)0, acc1, false, false);
    acc2 = __builtin_amdgcn_wmma_f32_16x16x32_f16(false, a, false, B2, (short)0, acc2, false, false);
    acc3 = __builtin_amdgcn_wmma_f32_16x16x32_f16(false, a, false, B3, (short)0, acc3, false, false);
  }

  const int mbase = tm * 16 + (half << 3);    // C/D layout: vgpr v -> M=mbase+v
  float* Cb = C + (size_t)mbase * N + (tg << 6) + l15;
#pragma unroll
  for (int v = 0; v < 8; ++v) {
    Cb[(size_t)v * N +  0] = acc0[v];
    Cb[(size_t)v * N + 16] = acc1[v];
    Cb[(size_t)v * N + 32] = acc2[v];
    Cb[(size_t)v * N + 48] = acc3[v];
  }
}

// ---------------------------------------------------------------------------
// out[dst] += dinv[src]*dinv[dst] * h[src]   (L2-resident gather + f32 atomics)
// thread -> (edge, 4 consecutive features); consecutive threads are coalesced.
// ---------------------------------------------------------------------------
__global__ void gcn_scatter(const float* __restrict__ h, const int* __restrict__ src,
                            const int* __restrict__ dst, const float* __restrict__ dinv,
                            float* __restrict__ out, int E, int Hn) {
  long long tid = (long long)blockIdx.x * blockDim.x + threadIdx.x;
  int chunks = Hn >> 2;
  long long e = tid / chunks;
  if (e >= E) return;
  int f = (int)(tid - e * chunks) << 2;
  int s = src[e], d = dst[e];
  float w = dinv[s] * dinv[d];
  const float4 hv = *reinterpret_cast<const float4*>(h + (size_t)s * Hn + f);
  float* o = out + (size_t)d * Hn + f;
  atomicAdd(o + 0, w * hv.x);
  atomicAdd(o + 1, w * hv.y);
  atomicAdd(o + 2, w * hv.z);
  atomicAdd(o + 3, w * hv.w);
}

// agg = BN( agg + dinv[i]^2 * h[i] (self loop) + bias ) [+ ReLU]
__global__ void bn_epilogue(float* __restrict__ agg, const float* __restrict__ h,
                            const float* __restrict__ dinv, const float* __restrict__ bias,
                            const float* __restrict__ g, const float* __restrict__ be,
                            const float* __restrict__ m, const float* __restrict__ v,
                            int n, int Hn, int relu) {
  long long tid = (long long)blockIdx.x * blockDim.x + threadIdx.x;
  if (tid >= (long long)n * Hn) return;
  int f = (int)(tid % Hn);
  int i = (int)(tid / Hn);
  float sl = dinv[i]; sl *= sl;
  float val = agg[tid] + sl * h[tid] + bias[f];
  val = g[f] * (val - m[f]) * rsqrtf(v[f] + 1e-5f) + be[f];
  if (relu) val = fmaxf(val, 0.0f);
  agg[tid] = val;
}

// segment mean-pool helpers
__global__ void seg_pool_sum(const float* __restrict__ x, const int* __restrict__ seg,
                             float* __restrict__ sums, float* __restrict__ cnt,
                             int n, int Hn) {
  long long tid = (long long)blockIdx.x * blockDim.x + threadIdx.x;
  if (tid >= (long long)n * Hn) return;
  int i = (int)(tid / Hn), f = (int)(tid % Hn);
  int sgi = seg[i];
  atomicAdd(&sums[(size_t)sgi * Hn + f], x[tid]);
  if (f == 0) atomicAdd(&cnt[sgi], 1.0f);
}

__global__ void pool_div(float* __restrict__ sums, const float* __restrict__ cnt,
                         int n, int Hn) {
  long long tid = (long long)blockIdx.x * blockDim.x + threadIdx.x;
  if (tid >= (long long)n * Hn) return;
  int i = (int)(tid / Hn);
  sums[tid] /= fmaxf(cnt[i], 1.0f);
}

__global__ void batch_sum(const int* __restrict__ batch, const int* __restrict__ pool,
                          float* __restrict__ sumb, int n) {
  int i = blockIdx.x * blockDim.x + threadIdx.x;
  if (i < n) atomicAdd(&sumb[pool[i]], (float)batch[i]);
}

__global__ void bpool_round(const float* __restrict__ sumb, const float* __restrict__ cnt,
                            int* __restrict__ bpool, int n) {
  int i = blockIdx.x * blockDim.x + threadIdx.x;
  if (i < n) bpool[i] = (int)rintf(sumb[i] / fmaxf(cnt[i], 1.0f));
}

// out = softmax(x @ Wl + bl) per row; optionally also write [probs, extra]
// to a concat buffer (row stride catStride; remaining cols stay zero).
__global__ void lin_softmax(const float* __restrict__ x, const float* __restrict__ Wl,
                            const float* __restrict__ bl, float* __restrict__ out,
                            float* __restrict__ cat, const float* __restrict__ extra,
                            int n, int Hn, int Cn, int catStride) {
  int i = blockIdx.x * blockDim.x + threadIdx.x;
  if (i >= n) return;
  float s[16];
  for (int c = 0; c < Cn; ++c) s[c] = bl[c];
  for (int k = 0; k < Hn; ++k) {
    float xv = x[(size_t)i * Hn + k];
    for (int c = 0; c < Cn; ++c) s[c] += xv * Wl[(size_t)k * Cn + c];
  }
  float mx = s[0];
  for (int c = 1; c < Cn; ++c) mx = fmaxf(mx, s[c]);
  float den = 0.0f;
  for (int c = 0; c < Cn; ++c) { s[c] = __expf(s[c] - mx); den += s[c]; }
  float inv = 1.0f / den;
  for (int c = 0; c < Cn; ++c) {
    float p = s[c] * inv;
    out[(size_t)i * Cn + c] = p;
    if (cat) cat[(size_t)i * catStride + c] = p;
  }
  if (cat) cat[(size_t)i * catStride + Cn] = extra[i];
}

// ---------------------------------------------------------------------------
extern "C" void kernel_launch(void* const* d_in, const int* in_sizes, int n_in,
                              void* d_out, int out_size, void* d_ws, size_t ws_size,
                              hipStream_t stream) {
  (void)in_sizes; (void)n_in; (void)out_size; (void)ws_size;
  const float* x     = (const float*)d_in[0];
  const float* xp1   = (const float*)d_in[1];
  const float* W_in0 = (const float*)d_in[2];
  const float* W_h0  = (const float*)d_in[3];
  const float* b0    = (const float*)d_in[4];
  const float* g0    = (const float*)d_in[5];
  const float* be0   = (const float*)d_in[6];
  const float* m0    = (const float*)d_in[7];
  const float* v0    = (const float*)d_in[8];
  const float* W_in1 = (const float*)d_in[9];
  const float* W_h1  = (const float*)d_in[10];
  const float* b1    = (const float*)d_in[11];
  const float* g1    = (const float*)d_in[12];
  const float* be1   = (const float*)d_in[13];
  const float* m1    = (const float*)d_in[14];
  const float* v1    = (const float*)d_in[15];
  const float* linW0 = (const float*)d_in[16];
  const float* linb0 = (const float*)d_in[17];
  const float* linW1 = (const float*)d_in[18];
  const float* linb1 = (const float*)d_in[19];
  const int* ei0   = (const int*)d_in[20];
  const int* batch = (const int*)d_in[21];
  const int* pool1 = (const int*)d_in[22];
  const int* ei1   = (const int*)d_in[23];
  const int *src0 = ei0, *dst0 = ei0 + E0c;
  const int *src1 = ei1, *dst1 = ei1 + E1c;
  float* out = (float*)d_out;

  // ---- workspace layout (all offsets multiples of 16 floats) ----
  float* ws = (float*)d_ws;
  size_t off = 0;
  float* bufA  = ws + off; off += (size_t)N0c * Hc;       // 7,680,000
  float* bufH  = ws + off; off += (size_t)N0c * Hc;
  float* bufC  = ws + off; off += (size_t)N0c * Hc;
  float* dinv0 = ws + off; off += N0c;                    // 30000 (mult of 16)
  float* dinv1 = ws + off; off += N1p;
  float* cnt   = ws + off; off += N1p;
  float* sums  = ws + off; off += (size_t)N1c * Hc;       // 768,000
  float* sumb  = ws + off; off += N1p;
  int*   bpool = (int*)(ws + off); off += N1p;
  float* xp    = ws + off; off += (size_t)N1p * 32;       // K padded 17 -> 32
  float* gsum  = ws + off; off += (size_t)Bc * Hc;
  float* gcnt  = ws + off; off += 16;
  // pre-swizzled f16 weights (each K*N halves; offsets stay 64B aligned)
  _Float16* swz = (_Float16*)(ws + off);
  _Float16* sw_in0 = swz;                       // 128*256
  _Float16* sw_h0a = sw_in0 + 128 * 256;        // 256*256
  _Float16* sw_h0b = sw_h0a + 256 * 256;
  _Float16* sw_in1 = sw_h0b + 256 * 256;        // 32*256 (padded K)
  _Float16* sw_h1a = sw_in1 + 32 * 256;
  _Float16* sw_h1b = sw_h1a + 256 * 256;

  const int TPB = 256;

  // ---- weight swizzle (tiny, once per call) ----
  swizzle_w<<<(128 * 256 + TPB - 1) / TPB, TPB, 0, stream>>>(W_in0, sw_in0, 128, 128, Hc);
  swizzle_w<<<(256 * 256 + TPB - 1) / TPB, TPB, 0, stream>>>(W_h0, sw_h0a, 256, 256, Hc);
  swizzle_w<<<(256 * 256 + TPB - 1) / TPB, TPB, 0, stream>>>(W_h0 + Hc * Hc, sw_h0b, 256, 256, Hc);
  swizzle_w<<<(32 * 256 + TPB - 1) / TPB, TPB, 0, stream>>>(W_in1, sw_in1, 32, Cc + 1, Hc);
  swizzle_w<<<(256 * 256 + TPB - 1) / TPB, TPB, 0, stream>>>(W_h1, sw_h1a, 256, 256, Hc);
  swizzle_w<<<(256 * 256 + TPB - 1) / TPB, TPB, 0, stream>>>(W_h1 + Hc * Hc, sw_h1b, 256, 256, Hc);

  // ---- level-0 gcn_norm ----
  hipMemsetAsync(dinv0, 0, N0c * sizeof(float), stream);
  deg_count<<<(E0c + TPB - 1) / TPB, TPB, 0, stream>>>(dst0, dinv0, E0c);
  deg_to_dinv<<<(N0c + TPB - 1) / TPB, TPB, 0, stream>>>(dinv0, N0c);

  // ---- level-0 GCN stack (M=30000, mult of 16 -> no guards) ----
  const _Float16* sws0[3] = {sw_in0, sw_h0a, sw_h0b};
  const int       Kp0[3]  = {F0c, Hc, Hc};
  float* aggs0[3] = {bufA, bufC, bufA};
  const float* Ain = x;
  for (int l = 0; l < 3; ++l) {
    float* agg = aggs0[l];
    int waves = (N0c / 16) * (Hc / 64);
    gemm_f16_wmma<<<(waves + 7) / 8, TPB, 0, stream>>>(Ain, sws0[l], bufH, N0c, Kp0[l], Hc);
    hipMemsetAsync(agg, 0, (size_t)N0c * Hc * sizeof(float), stream);
    long long st = (long long)E0c * (Hc / 4);
    gcn_scatter<<<(int)((st + TPB - 1) / TPB), TPB, 0, stream>>>(bufH, src0, dst0, dinv0, agg, E0c, Hc);
    long long nt = (long long)N0c * Hc;
    bn_epilogue<<<(int)((nt + TPB - 1) / TPB), TPB, 0, stream>>>(
        agg, bufH, dinv0, b0 + l * Hc, g0 + l * Hc, be0 + l * Hc,
        m0 + l * Hc, v0 + l * Hc, N0c, Hc, l < 2);
    Ain = agg;
  }

  // ---- pool to clusters, cluster classifier, concat (K-padded to 32) ----
  hipMemsetAsync(sums, 0, (size_t)N1c * Hc * sizeof(float), stream);
  hipMemsetAsync(cnt, 0, N1p * sizeof(float), stream);
  hipMemsetAsync(sumb, 0, N1p * sizeof(float), stream);
  hipMemsetAsync(xp, 0, (size_t)N1p * 32 * sizeof(float), stream);  // K + row pad = 0
  long long pt = (long long)N0c * Hc;
  seg_pool_sum<<<(int)((pt + TPB - 1) / TPB), TPB, 0, stream>>>(Ain, pool1, sums, cnt, N0c, Hc);
  pool_div<<<(int)(((long long)N1c * Hc + TPB - 1) / TPB), TPB, 0, stream>>>(sums, cnt, N1c, Hc);
  batch_sum<<<(N0c + TPB - 1) / TPB, TPB, 0, stream>>>(batch, pool1, sumb, N0c);
  bpool_round<<<(N1c + TPB - 1) / TPB, TPB, 0, stream>>>(sumb, cnt, bpool, N1c);
  lin_softmax<<<(N1c + 63) / 64, 64, 0, stream>>>(sums, linW0, linb0, out, xp, xp1,
                                                  N1c, Hc, Cc, 32);

  // ---- pooled-graph gcn_norm ----
  hipMemsetAsync(dinv1, 0, N1p * sizeof(float), stream);
  deg_count<<<(E1c + TPB - 1) / TPB, TPB, 0, stream>>>(dst1, dinv1, E1c);
  deg_to_dinv<<<(N1c + TPB - 1) / TPB, TPB, 0, stream>>>(dinv1, N1c);

  // ---- pooled GCN stack (M padded to 3008 -> guard-free WMMA loop) ----
  const _Float16* sws1[3] = {sw_in1, sw_h1a, sw_h1b};
  const int       Kp1[3]  = {32, Hc, Hc};
  float* aggs1[3] = {bufA, bufC, bufA};   // reuse big buffers (>= 3008 rows)
  const float* A1 = xp;
  for (int l = 0; l < 3; ++l) {
    float* agg = aggs1[l];
    int waves = (N1p / 16) * (Hc / 64);
    gemm_f16_wmma<<<(waves + 7) / 8, TPB, 0, stream>>>(A1, sws1[l], bufH, N1p, Kp1[l], Hc);
    hipMemsetAsync(agg, 0, (size_t)N1p * Hc * sizeof(float), stream);  // pad rows stay 0
    long long st = (long long)E1c * (Hc / 4);
    gcn_scatter<<<(int)((st + TPB - 1) / TPB), TPB, 0, stream>>>(bufH, src1, dst1, dinv1, agg, E1c, Hc);
    long long nt = (long long)N1c * Hc;
    bn_epilogue<<<(int)((nt + TPB - 1) / TPB), TPB, 0, stream>>>(
        agg, bufH, dinv1, b1 + l * Hc, g1 + l * Hc, be1 + l * Hc,
        m1 + l * Hc, v1 + l * Hc, N1c, Hc, l < 2);
    A1 = agg;
  }

  // ---- final graph-level pool + classifier ----
  hipMemsetAsync(gsum, 0, (size_t)Bc * Hc * sizeof(float), stream);
  hipMemsetAsync(gcnt, 0, Bc * sizeof(float), stream);
  long long ft = (long long)N1c * Hc;
  seg_pool_sum<<<(int)((ft + TPB - 1) / TPB), TPB, 0, stream>>>(A1, bpool, gsum, gcnt, N1c, Hc);
  pool_div<<<((Bc * Hc) + TPB - 1) / TPB, TPB, 0, stream>>>(gsum, gcnt, Bc, Hc);
  lin_softmax<<<1, 64, 0, stream>>>(gsum, linW1, linb1, out + (size_t)N1c * Cc,
                                    nullptr, nullptr, Bc, Hc, Cc, 0);
}
